// GraphormerEncoderLayer_46437186404985
// MI455X (gfx1250) — matrix operand order repaired and checked by reading, not compiled, not run on hardware
//
#include <hip/hip_runtime.h>
#include <hip/hip_bf16.h>
#include <stdint.h>

#define NN 4096
#define CC 256
#define HH 8
#define DD 32
#define FF 512
#define WPR 128   // bitset words per row = NN/32

typedef __attribute__((ext_vector_type(16))) __bf16 v16bf;
typedef __attribute__((ext_vector_type(8)))  float  v8f;
typedef __attribute__((ext_vector_type(4)))  unsigned int v4u;
typedef __attribute__((ext_vector_type(8)))  int    v8i;
typedef __attribute__((ext_vector_type(4)))  int    v4i;

struct ushort16 { unsigned short s[16]; };

static __device__ __forceinline__ v16bf as_v16bf(ushort16 x) {
  return __builtin_bit_cast(v16bf, x);
}
static __device__ __forceinline__ unsigned short f2bf(float f) {
  unsigned int u = __builtin_bit_cast(unsigned int, f);
  u += 0x7FFFu + ((u >> 16) & 1u);     // round-to-nearest-even
  return (unsigned short)(u >> 16);
}
static __device__ __forceinline__ v8f wmma_bf16(v16bf a, v16bf b, v8f c) {
  return __builtin_amdgcn_wmma_f32_16x16x32_bf16(false, a, false, b, (short)0, c,
                                                 false, false);
}
// Load one A-fragment (two 8-elem chunks) and one B-fragment (16 contiguous).
static __device__ __forceinline__ void load_frag(const unsigned short* __restrict__ ar,
                                                 const unsigned short* __restrict__ br,
                                                 ushort16& a, ushort16& b) {
#pragma unroll
  for (int i = 0; i < 8; ++i) { a.s[i] = ar[i]; a.s[8 + i] = ar[8 + i]; }
#pragma unroll
  for (int i = 0; i < 16; ++i) b.s[i] = br[i];
}

// Issue one TDM load of a 2-D bf16 tile [32 rows x 32 cols] from global (row
// stride CC elements) into LDS at ldsoff, with 1-DWORD padding every 16 DWORDs
// (=> LDS row stride 34 ushorts, matching vlds[.][32][34]).
static __device__ __forceinline__ void tdm_load_v_tile(const unsigned short* gptr,
                                                       unsigned ldsoff) {
  unsigned long long ga = (unsigned long long)(uintptr_t)gptr;
  v4u g0;
  g0[0] = 1u;                                   // count=1, user mode
  g0[1] = ldsoff;                               // lds_addr (bytes)
  g0[2] = (unsigned)(ga & 0xFFFFFFFFu);         // global_addr[31:0]
  g0[3] = (unsigned)((ga >> 32) & 0x01FFFFFFu)  // global_addr[56:32]
          | 0x80000000u;                        // type=2 (bits 127:126 = 10b)
  v8i g1;
  g1[0] = (int)0x00D10000;      // data_size=1(2B) | pad_enable | pad_interval=3
  g1[1] = (int)(256u << 16);    // tensor_dim0 = 256 (bits 79:48)
  g1[2] = (int)(4096u << 16);   // tensor_dim1 = 4096 (bits 111:80)
  g1[3] = (int)(32u << 16);     // tile_dim0 = 32 (bits 127:112)
  g1[4] = (int)32;              // tile_dim1 = 32, tile_dim2 = 0
  g1[5] = (int)256;             // tensor_dim0_stride = 256 (bits 207:160)
  g1[6] = 0;
  g1[7] = 0;
  v4i gz = {0, 0, 0, 0};
#if defined(__clang_major__) && (__clang_major__ >= 23)
  v8i gz8 = {0, 0, 0, 0, 0, 0, 0, 0};
  __builtin_amdgcn_tensor_load_to_lds(g0, g1, gz, gz, gz8, 0);
#else
  __builtin_amdgcn_tensor_load_to_lds(g0, g1, gz, gz, 0);
#endif
}

// ---------------- degree + hidden ----------------
__global__ void deg_kernel(const int* __restrict__ ei, int* __restrict__ deg, int E) {
  int e = blockIdx.x * blockDim.x + threadIdx.x;
  if (e < E) {
    atomicAdd(&deg[ei[e]], 1);
    atomicAdd(&deg[ei[E + e]], 1);
  }
}

__global__ void hidden_kernel(const float* __restrict__ x, const int* __restrict__ deg,
                              const float* __restrict__ deg_emb,
                              unsigned short* __restrict__ hb) {
  int n = blockIdx.x, t = threadIdx.x;
  int d = deg[n]; if (d > 33) d = 33;
  float h = x[(size_t)n * CC + t] + deg_emb[(size_t)d * CC + t];
  hb[(size_t)n * CC + t] = f2bf(h);
}

// ---------------- weight transpose to bf16 ----------------
__global__ void wtrans_kernel(const float* __restrict__ W, unsigned short* __restrict__ WT,
                              int K, int Nc) {
  int n = blockIdx.x;
  for (int k = threadIdx.x; k < K; k += blockDim.x)
    WT[(size_t)n * K + k] = f2bf(W[(size_t)k * Nc + n]);
}

// ---------------- generic WMMA GEMM: out = A[M,K] @ W[K,Nc] + bias ----------------
// K is a compile-time constant (256 or 512): the K loop fully unrolls, each
// iteration gets its own fragment registers (no rotation copies), and all
// global_load_b128s can be hoisted ahead of the serialized WMMA chain.
template <int K>
__global__ void gemm_bf16_wmma(const unsigned short* __restrict__ A,
                               const unsigned short* __restrict__ WT,
                               const float* __restrict__ bias,
                               float* __restrict__ outf,
                               unsigned short* __restrict__ outb,
                               int Nc, int act) {
  int n0 = blockIdx.x * 16;
  int m0 = blockIdx.y * 16;
  int lane = threadIdx.x;
  int lr = lane & 15, lh = lane >> 4;
  const unsigned short* arow = A + (size_t)(m0 + lr) * K + lh * 16;
  const unsigned short* brow = WT + (size_t)(n0 + lr) * K + lh * 16;
  v8f acc = {};
#pragma unroll
  for (int k = 0; k < K; k += 32) {
    ushort16 a, b;
    load_frag(arow + k, brow + k, a, b);
    acc = wmma_bf16(as_v16bf(a), as_v16bf(b), acc);
  }
  float bv = bias ? bias[n0 + lr] : 0.0f;
#pragma unroll
  for (int r = 0; r < 8; ++r) {
    float xv = acc[r] + bv;
    if (act == 1) xv = 0.5f * xv * (1.0f + erff(xv * 0.70710678118654752f));
    int row = m0 + r + 8 * lh;
    size_t o = (size_t)row * Nc + n0 + lr;
    if (outf) outf[o] = xv;
    if (outb) outb[o] = f2bf(xv);
  }
}

// ---------------- BFS distance buckets (bitset) ----------------
__global__ void adj_kernel(const int* __restrict__ ei, unsigned int* __restrict__ adj, int E) {
  int e = blockIdx.x * blockDim.x + threadIdx.x;
  if (e < E) {
    int r = ei[e], c = ei[E + e];
    atomicOr(&adj[(size_t)r * WPR + (c >> 5)], 1u << (c & 31));
    atomicOr(&adj[(size_t)c * WPR + (r >> 5)], 1u << (r & 31));
  }
}

__global__ void dist_init_kernel(unsigned int* __restrict__ dist4, size_t nwords) {
  size_t i = (size_t)blockIdx.x * blockDim.x + threadIdx.x;
  if (i < nwords) dist4[i] = 0x05050505u;  // far = MAX_DIST+1 = 5
}

__global__ void dist_diag_kernel(unsigned char* __restrict__ dist) {
  int n = blockIdx.x * blockDim.x + threadIdx.x;
  if (n < NN) dist[(size_t)n * NN + n] = 0;
}

__global__ void bfs_init_kernel(const unsigned int* __restrict__ adj,
                                unsigned int* __restrict__ front,
                                unsigned int* __restrict__ seen) {
  int n = blockIdx.x, t = threadIdx.x;   // 128 threads
  size_t idx = (size_t)n * WPR + t;
  front[idx] = adj[idx];
  seen[idx] = (t == (n >> 5)) ? (1u << (n & 31)) : 0u;
}

__global__ void bfs_mark_kernel(unsigned int* __restrict__ front,
                                unsigned int* __restrict__ seen,
                                unsigned char* __restrict__ dist, int d) {
  int n = blockIdx.x, t = threadIdx.x;   // 128 threads
  size_t idx = (size_t)n * WPR + t;
  unsigned int f = front[idx] & ~seen[idx];
  front[idx] = f;
  seen[idx] |= f;
  unsigned char* dr = dist + (size_t)n * NN + t * 32;
  while (f) {
    int b = __builtin_ctz(f);
    f &= f - 1;
    dr[b] = (unsigned char)d;
  }
}

__global__ void bfs_expand_kernel(const unsigned int* __restrict__ front,
                                  const unsigned int* __restrict__ adj,
                                  unsigned int* __restrict__ next) {
  __shared__ unsigned int fr[WPR];
  int n = blockIdx.x, t = threadIdx.x;   // 128 threads
  fr[t] = front[(size_t)n * WPR + t];
  __syncthreads();
  unsigned int acc = 0;
  for (int w = 0; w < WPR; ++w) {
    unsigned int bits = fr[w];
    while (bits) {
      int b = __builtin_ctz(bits);
      bits &= bits - 1;
      acc |= adj[(size_t)(w * 32 + b) * WPR + t];
    }
  }
  next[(size_t)n * WPR + t] = acc;
}

// ---- fused attention (flash-style, WMMA + double-buffered TDM V staging) ----
__global__ void attn_wmma_kernel(const unsigned short* __restrict__ q,
                                 const unsigned short* __restrict__ k,
                                 const unsigned short* __restrict__ v,
                                 const unsigned char* __restrict__ dist,
                                 const float* __restrict__ spa,   // [6,H]
                                 unsigned short* __restrict__ outb) {
  __shared__ unsigned short plds[16][33];
  __shared__ unsigned short vlds[2][32][34]; // TDM pads rows: 16 DW + 1 DW pad
  int q0 = blockIdx.x * 16;
  int head = blockIdx.y;
  int lane = threadIdx.x, lr = lane & 15, lh = lane >> 4;
  const float scale = 0.17677669529663687f;   // 1/sqrt(32)
  float tab[6];
#pragma unroll
  for (int i = 0; i < 6; ++i) tab[i] = spa[i * HH + head];
  unsigned vbase = (unsigned)(uintptr_t)(&vlds[0][0][0]);
  const unsigned vstride = 32 * 34 * 2;       // bytes per buffer

  // Q fragment (A layout), reused all iterations
  ushort16 uq;
  {
    const unsigned short* qp = q + (size_t)(q0 + lr) * CC + head * DD;
    int base = lh * 16;
#pragma unroll
    for (int i = 0; i < 8; ++i) {
      uq.s[i]     = qp[base + i];
      uq.s[8 + i] = qp[base + 8 + i];
    }
  }
  v16bf aq = as_v16bf(uq);

  v8f acc0 = {}, acc1 = {};
  float mrow[8], lsum[8];
#pragma unroll
  for (int r = 0; r < 8; ++r) { mrow[r] = -1e30f; lsum[r] = 0.0f; }

  // prologue: DMA the first V block
  tdm_load_v_tile(v + head * DD, vbase);
  int cur = 0;

  for (int m = 0; m < NN; m += 32) {
    bool more = (m + 32 < NN);                 // wave-uniform
    if (more)
      tdm_load_v_tile(v + (size_t)(m + 32) * CC + head * DD,
                      vbase + (unsigned)(cur ^ 1) * vstride);
    if (more)
      __builtin_prefetch(k + (size_t)(m + 32 + lr) * CC + head * DD, 0, 1);

    // scores: two 16x16 tiles, K = D = 32 (overlaps with the TDM transfers)
    v8f s0, s1;
    {
      ushort16 ub;
      const unsigned short* kp = k + (size_t)(m + lr) * CC + head * DD + lh * 16;
#pragma unroll
      for (int i = 0; i < 16; ++i) ub.s[i] = kp[i];
      v8f z = {};
      s0 = wmma_bf16(aq, as_v16bf(ub), z);
      kp += 16 * CC;
#pragma unroll
      for (int i = 0; i < 16; ++i) ub.s[i] = kp[i];
      s1 = wmma_bf16(aq, as_v16bf(ub), z);
    }

    // bias + online softmax; P tile to LDS (D-layout write)
    int row_off = q0 + 8 * lh;
#pragma unroll
    for (int r = 0; r < 8; ++r) {
      const unsigned char* dr = dist + (size_t)(row_off + r) * NN + m;
      float x0 = s0[r] * scale + tab[dr[lr]];
      float x1 = s1[r] * scale + tab[dr[16 + lr]];
      float mc = fmaxf(x0, x1);
      mc = fmaxf(mc, __shfl_xor(mc, 1, 32));
      mc = fmaxf(mc, __shfl_xor(mc, 2, 32));
      mc = fmaxf(mc, __shfl_xor(mc, 4, 32));
      mc = fmaxf(mc, __shfl_xor(mc, 8, 32));
      float mn = fmaxf(mrow[r], mc);
      float corr = __expf(mrow[r] - mn);
      mrow[r] = mn;
      float p0 = __expf(x0 - mn);
      float p1 = __expf(x1 - mn);
      lsum[r] = lsum[r] * corr + p0 + p1;
      acc0[r] *= corr;
      acc1[r] *= corr;
      plds[r + 8 * lh][lr]      = f2bf(p0);
      plds[r + 8 * lh][16 + lr] = f2bf(p1);
    }
    __syncthreads();

    // P fragment (A layout)
    ushort16 up, uv0, uv1;
    {
      int base = lh * 16;
#pragma unroll
      for (int i = 0; i < 8; ++i) {
        up.s[i]     = plds[lr][base + i];
        up.s[8 + i] = plds[lr][base + 8 + i];
      }
    }
    // V fragments (B layout) — current tile guaranteed complete when at most
    // one (the prefetched next) TDM op is still outstanding (in-order counter).
    if (more) __builtin_amdgcn_s_wait_tensorcnt(1);
    else      __builtin_amdgcn_s_wait_tensorcnt(0);
#pragma unroll
    for (int t = 0; t < 16; ++t) {
      uv0.s[t] = vlds[cur][lh * 16 + t][lr];
      uv1.s[t] = vlds[cur][lh * 16 + t][16 + lr];
    }
    v16bf ap = as_v16bf(up);
    acc0 = wmma_bf16(ap, as_v16bf(uv0), acc0);
    acc1 = wmma_bf16(ap, as_v16bf(uv1), acc1);
    __syncthreads();
    cur ^= 1;
  }

  // finalize: divide by row sum, write bf16
#pragma unroll
  for (int r = 0; r < 8; ++r) {
    float s = lsum[r];
    s += __shfl_xor(s, 1, 32);
    s += __shfl_xor(s, 2, 32);
    s += __shfl_xor(s, 4, 32);
    s += __shfl_xor(s, 8, 32);
    float inv = 1.0f / s;
    size_t o = (size_t)(q0 + r + 8 * lh) * CC + head * DD;
    outb[o + lr]      = f2bf(acc0[r] * inv);
    outb[o + 16 + lr] = f2bf(acc1[r] * inv);
  }
}

// ---------------- residual + LayerNorm ----------------
__global__ void ln_residual_kernel(const float* __restrict__ a, const float* __restrict__ b,
                                   const float* __restrict__ gamma, const float* __restrict__ beta,
                                   float* __restrict__ outf, unsigned short* __restrict__ outb) {
  __shared__ float red[CC];
  int n = blockIdx.x, t = threadIdx.x;   // 256 threads
  size_t o = (size_t)n * CC + t;
  float x = a[o] + b[o];
  red[t] = x;
  __syncthreads();
  for (int s = CC / 2; s > 0; s >>= 1) {
    if (t < s) red[t] += red[t + s];
    __syncthreads();
  }
  float mu = red[0] * (1.0f / CC);
  __syncthreads();
  float d = x - mu;
  red[t] = d * d;
  __syncthreads();
  for (int s = CC / 2; s > 0; s >>= 1) {
    if (t < s) red[t] += red[t + s];
    __syncthreads();
  }
  float var = red[0] * (1.0f / CC);
  float y = d * rsqrtf(var + 1e-5f) * gamma[t] + beta[t];
  if (outf) outf[o] = y;
  if (outb) outb[o] = f2bf(y);
}

// ---------------- launch ----------------
extern "C" void kernel_launch(void* const* d_in, const int* in_sizes, int n_in,
                              void* d_out, int out_size, void* d_ws, size_t ws_size,
                              hipStream_t stream) {
  (void)n_in; (void)out_size; (void)ws_size;
  const float* x       = (const float*)d_in[0];
  const int*   ei      = (const int*)d_in[1];
  const float* deg_emb = (const float*)d_in[2];
  const float* spa_emb = (const float*)d_in[3];
  const float* Wq = (const float*)d_in[4];  const float* bq = (const float*)d_in[5];
  const float* Wk = (const float*)d_in[6];  const float* bk = (const float*)d_in[7];
  const float* Wv = (const float*)d_in[8];  const float* bv = (const float*)d_in[9];
  const float* Wo = (const float*)d_in[10]; const float* bo = (const float*)d_in[11];
  const float* g1 = (const float*)d_in[12]; const float* b1 = (const float*)d_in[13];
  const float* g2 = (const float*)d_in[14]; const float* b2 = (const float*)d_in[15];
  const float* Wf1 = (const float*)d_in[16]; const float* bf1 = (const float*)d_in[17];
  const float* Wf2 = (const float*)d_in[18]; const float* bf2 = (const float*)d_in[19];
  float* out = (float*)d_out;
  const int E = in_sizes[1] / 2;

  char* p = (char*)d_ws;
  auto carve = [&](size_t bytes) -> void* {
    void* r = (void*)p;
    p += (bytes + 255) & ~(size_t)255;
    return r;
  };
  int*            deg      = (int*)carve((size_t)NN * 4);
  unsigned short* hidden_b = (unsigned short*)carve((size_t)NN * CC * 2);
  unsigned short* qb       = (unsigned short*)carve((size_t)NN * CC * 2);
  unsigned short* kb       = (unsigned short*)carve((size_t)NN * CC * 2);
  unsigned short* vb       = (unsigned short*)carve((size_t)NN * CC * 2);
  unsigned short* att_b    = (unsigned short*)carve((size_t)NN * CC * 2);
  float*          attnout  = (float*)carve((size_t)NN * CC * 4);
  float*          h_f      = (float*)carve((size_t)NN * CC * 4);
  unsigned short* h_b      = (unsigned short*)carve((size_t)NN * CC * 2);
  unsigned short* ff1b     = (unsigned short*)carve((size_t)NN * FF * 2);
  float*          ff2out   = (float*)carve((size_t)NN * CC * 4);
  unsigned short* WTq      = (unsigned short*)carve((size_t)CC * CC * 2);
  unsigned short* WTk      = (unsigned short*)carve((size_t)CC * CC * 2);
  unsigned short* WTv      = (unsigned short*)carve((size_t)CC * CC * 2);
  unsigned short* WTo      = (unsigned short*)carve((size_t)CC * CC * 2);
  unsigned short* WTf1     = (unsigned short*)carve((size_t)FF * CC * 2);
  unsigned short* WTf2     = (unsigned short*)carve((size_t)CC * FF * 2);
  unsigned int*   adj      = (unsigned int*)carve((size_t)NN * WPR * 4);
  unsigned int*   seen     = (unsigned int*)carve((size_t)NN * WPR * 4);
  unsigned int*   fA       = (unsigned int*)carve((size_t)NN * WPR * 4);
  unsigned int*   fB       = (unsigned int*)carve((size_t)NN * WPR * 4);
  unsigned char*  dist     = (unsigned char*)carve((size_t)NN * NN);

  hipMemsetAsync(deg, 0, (size_t)NN * 4, stream);
  hipMemsetAsync(adj, 0, (size_t)NN * WPR * 4, stream);

  deg_kernel<<<(E + 255) / 256, 256, 0, stream>>>(ei, deg, E);
  hidden_kernel<<<NN, CC, 0, stream>>>(x, deg, deg_emb, hidden_b);

  wtrans_kernel<<<CC, 256, 0, stream>>>(Wq, WTq, CC, CC);
  wtrans_kernel<<<CC, 256, 0, stream>>>(Wk, WTk, CC, CC);
  wtrans_kernel<<<CC, 256, 0, stream>>>(Wv, WTv, CC, CC);
  wtrans_kernel<<<CC, 256, 0, stream>>>(Wo, WTo, CC, CC);
  wtrans_kernel<<<FF, 256, 0, stream>>>(Wf1, WTf1, CC, FF);
  wtrans_kernel<<<CC, 256, 0, stream>>>(Wf2, WTf2, FF, CC);

  dim3 gqkv(CC / 16, NN / 16);
  gemm_bf16_wmma<CC><<<gqkv, 32, 0, stream>>>(hidden_b, WTq, bq, nullptr, qb, CC, 0);
  gemm_bf16_wmma<CC><<<gqkv, 32, 0, stream>>>(hidden_b, WTk, bk, nullptr, kb, CC, 0);
  gemm_bf16_wmma<CC><<<gqkv, 32, 0, stream>>>(hidden_b, WTv, bv, nullptr, vb, CC, 0);

  adj_kernel<<<(E + 255) / 256, 256, 0, stream>>>(ei, adj, E);
  dist_init_kernel<<<(NN * NN / 4 + 255) / 256, 256, 0, stream>>>((unsigned int*)dist,
                                                                  (size_t)NN * NN / 4);
  dist_diag_kernel<<<NN / 256, 256, 0, stream>>>(dist);
  bfs_init_kernel<<<NN, WPR, 0, stream>>>(adj, fA, seen);
  bfs_mark_kernel<<<NN, WPR, 0, stream>>>(fA, seen, dist, 1);
  bfs_expand_kernel<<<NN, WPR, 0, stream>>>(fA, adj, fB);
  bfs_mark_kernel<<<NN, WPR, 0, stream>>>(fB, seen, dist, 2);
  bfs_expand_kernel<<<NN, WPR, 0, stream>>>(fB, adj, fA);
  bfs_mark_kernel<<<NN, WPR, 0, stream>>>(fA, seen, dist, 3);
  bfs_expand_kernel<<<NN, WPR, 0, stream>>>(fA, adj, fB);
  bfs_mark_kernel<<<NN, WPR, 0, stream>>>(fB, seen, dist, 4);

  attn_wmma_kernel<<<dim3(NN / 16, HH), 32, 0, stream>>>(qb, kb, vb, dist, spa_emb, att_b);

  gemm_bf16_wmma<CC><<<gqkv, 32, 0, stream>>>(att_b, WTo, bo, attnout, nullptr, CC, 0);
  ln_residual_kernel<<<NN, CC, 0, stream>>>(x, attnout, g1, b1, h_f, h_b);

  gemm_bf16_wmma<CC><<<dim3(FF / 16, NN / 16), 32, 0, stream>>>(h_b, WTf1, bf1, nullptr,
                                                                ff1b, FF, 1);
  gemm_bf16_wmma<FF><<<gqkv, 32, 0, stream>>>(ff1b, WTf2, bf2, ff2out, nullptr, CC, 0);
  ln_residual_kernel<<<NN, CC, 0, stream>>>(h_f, ff2out, g2, b2, out, nullptr);
}
